// MultiHeadAttention_56873956933975
// MI455X (gfx1250) — compile-verified, matrix operand order checked by reading, not run on hardware
//
#include <hip/hip_runtime.h>

#define S_LEN  1024
#define DMODEL 1024
#define NHEAD  16
#define DK     64

typedef __attribute__((ext_vector_type(16))) __bf16          v16bf;
typedef __attribute__((ext_vector_type(8)))  float           v8f;
typedef __attribute__((ext_vector_type(8)))  unsigned short  ushort8;

__device__ __forceinline__ unsigned short f2bf(float f) {
  union { float f; unsigned int u; } x; x.f = f;
  unsigned int lsb = (x.u >> 16) & 1u;
  return (unsigned short)((x.u + 0x7FFFu + lsb) >> 16);  // round-to-nearest-even
}

// Async global->LDS 128-bit copy (CDNA5 GLOBAL_LOAD_ASYNC_TO_LDS_B128,
// GVS mode: mem = SGPR64 + VGPR32 + 0). Tracked by ASYNCcnt.
__device__ __forceinline__ void async_copy_b128(unsigned int lds_byte_off,
                                                const unsigned short* sbase,
                                                unsigned int gbyte_off) {
  asm volatile("global_load_async_to_lds_b128 %0, %1, %2"
               :: "v"(lds_byte_off), "v"(gbyte_off), "s"(sbase)
               : "memory");
}
__device__ __forceinline__ void wait_async0() {
  asm volatile("s_wait_asynccnt 0x0" ::: "memory");
}
__device__ __forceinline__ unsigned int lds_off(const void* p) {
  return (unsigned int)(unsigned long long)p;  // low 32 bits = LDS offset
}

// Load one 16-lane-striped bf16 fragment (A 16x32 row=M, or B 32x16 row=N).
// lane half=0: K = kbase+{0..7,16..23}; half=1: K = kbase+{8..15,24..31}.
__device__ __forceinline__ v16bf load_frag(const unsigned short* base, int row,
                                           int stride, int kbase, int half) {
  union { v16bf bf; ushort8 h[2]; } f;
  const unsigned short* p = base + (size_t)row * stride + kbase + half * 8;
  f.h[0] = *(const ushort8*)(p);
  f.h[1] = *(const ushort8*)(p + 16);
  return f.bf;
}

__device__ __forceinline__ v8f wmma_bf16(v16bf a, v16bf b, v8f c) {
  return __builtin_amdgcn_wmma_f32_16x16x32_bf16(false, a, false, b,
                                                 (short)0, c, false, false);
}

// ---------------------------------------------------------------------------
// One-time weight prep: Wt_bf16[n][k] = bf16(W[k][n])  (transpose + convert)
// ---------------------------------------------------------------------------
__global__ __launch_bounds__(256)
void wprep_kernel(const float* __restrict__ W, unsigned short* __restrict__ Wt) {
  __shared__ float t[32][33];
  const int bx = blockIdx.x * 32;   // n base
  const int by = blockIdx.y * 32;   // k base
  const int lx = threadIdx.x & 31, ly = threadIdx.x >> 5;  // 32 x 8
#pragma unroll
  for (int i = 0; i < 32; i += 8)
    t[ly + i][lx] = W[(size_t)(by + ly + i) * DMODEL + bx + lx];
  __syncthreads();
#pragma unroll
  for (int i = 0; i < 32; i += 8)
    Wt[(size_t)(bx + ly + i) * DMODEL + by + lx] = f2bf(t[lx][ly + i]);
}

// ---------------------------------------------------------------------------
// Projection GEMM: out = bf16(X[8192,1024] @ W + bias), W given as bf16 W^T.
// Workgroup: 256 threads (8 waves), tile 64(M) x 256(N), K-step 64.
// vmode=1: store transposed per-head layout out[((b*H+h)*DK+dv)*S + s].
// ---------------------------------------------------------------------------
__global__ __launch_bounds__(256)
void proj_bf16_kernel(const float* __restrict__ X,
                      const unsigned short* __restrict__ Wt,
                      const float* __restrict__ bias,
                      unsigned short* __restrict__ out, int vmode) {
  __shared__ unsigned short lA[64 * 64];    // X tile [m][k], bf16 (8KB)
  __shared__ unsigned short lB[256 * 64];   // W^T tile [n][k], bf16 (32KB)

  const int tid    = threadIdx.x;
  const int m_base = blockIdx.y * 64;
  const int n_base = blockIdx.x * 256;
  const int lane   = tid & 31, wv = tid >> 5;
  const int ln     = lane & 15, half = lane >> 4;
  const int wm     = (wv >> 2) * 32;
  const int wn     = (wv & 3) * 64;

  v8f acc[2][4];
#pragma unroll
  for (int i = 0; i < 2; ++i)
#pragma unroll
    for (int j = 0; j < 4; ++j) acc[i][j] = {0, 0, 0, 0, 0, 0, 0, 0};

  for (int k0 = 0; k0 < DMODEL; k0 += 64) {
    // stage W^T tile via async copies: 2048 x b128 across 256 threads
#pragma unroll
    for (int i = 0; i < 8; ++i) {
      int fid = tid + i * 256;
      int row = fid >> 3, c = fid & 7;       // row: n in [0,256), c: 8-elt chunk
      async_copy_b128(lds_off(&lB[row * 64 + c * 8]), Wt,
                      (unsigned)(((n_base + row) * DMODEL + k0 + c * 8) * 2));
    }
    // stage + convert X tile 64x64 f32 -> bf16 (1024 float4)
#pragma unroll
    for (int i = 0; i < 4; ++i) {
      int fid = tid + i * 256;
      int row = fid >> 4, c4 = fid & 15;
      float4 v = *(const float4*)(X + (size_t)(m_base + row) * DMODEL + k0 + c4 * 4);
      unsigned short* d = &lA[row * 64 + c4 * 4];
      d[0] = f2bf(v.x); d[1] = f2bf(v.y); d[2] = f2bf(v.z); d[3] = f2bf(v.w);
    }
    wait_async0();
    __syncthreads();

#pragma unroll
    for (int ks = 0; ks < 64; ks += 32) {
      v16bf a0 = load_frag(lA, wm + ln,      64, ks, half);
      v16bf a1 = load_frag(lA, wm + 16 + ln, 64, ks, half);
#pragma unroll
      for (int nt = 0; nt < 4; ++nt) {
        v16bf bb = load_frag(lB, wn + nt * 16 + ln, 64, ks, half);
        acc[0][nt] = wmma_bf16(a0, bb, acc[0][nt]);
        acc[1][nt] = wmma_bf16(a1, bb, acc[1][nt]);
      }
    }
    __syncthreads();
  }

  // epilogue: + bias, convert, store (optionally per-head transposed for V)
#pragma unroll
  for (int mt = 0; mt < 2; ++mt)
#pragma unroll
    for (int nt = 0; nt < 4; ++nt) {
      int ncol = n_base + wn + nt * 16 + ln;
      float bv = bias[ncol];
#pragma unroll
      for (int r = 0; r < 8; ++r) {
        int mrow = m_base + wm + mt * 16 + r + 8 * half;
        unsigned short val = f2bf(acc[mt][nt][r] + bv);
        if (vmode) {
          int bb2 = mrow >> 10, s = mrow & (S_LEN - 1);
          int hh = ncol >> 6, dv = ncol & (DK - 1);
          out[(size_t)(((bb2 * NHEAD + hh) * DK) + dv) * S_LEN + s] = val;
        } else {
          out[(size_t)mrow * DMODEL + ncol] = val;
        }
      }
    }
}

// ---------------------------------------------------------------------------
// Flash attention: per (b, h, 64-query block); 4 waves x 16 query rows.
// 64-key rounds; K staged [key][dk], V staged from pre-transposed V^T global.
// ---------------------------------------------------------------------------
__global__ __launch_bounds__(128)
void flash_attn_kernel(const unsigned short* __restrict__ Qw,
                       const unsigned short* __restrict__ Kw,
                       const unsigned short* __restrict__ VTw,
                       float* __restrict__ out) {
  __shared__ unsigned short Kt[64 * 64];     // [key][dk]   (8KB)
  __shared__ unsigned short Vt[64 * 64];     // [dv][key]   (8KB)
  __shared__ unsigned short Pl[4 * 16 * 64]; // per-wave P  (8KB)

  const int tid  = threadIdx.x;
  const int lane = tid & 31, wv = tid >> 5;
  const int ln   = lane & 15, half = lane >> 4;
  const int b = blockIdx.z, h = blockIdx.y;
  const int qbase = blockIdx.x * 64;

  const size_t head_off = ((size_t)b * S_LEN) * DMODEL + (size_t)h * DK;
  const unsigned short* Qb  = Qw + head_off;
  const unsigned short* Kb  = Kw + head_off;
  const unsigned short* VTb = VTw + (size_t)((b * NHEAD + h) * DK) * S_LEN;

  const int qrow = qbase + wv * 16 + ln;
  v16bf aq0 = load_frag(Qb, qrow, DMODEL, 0,  half);
  v16bf aq1 = load_frag(Qb, qrow, DMODEL, 32, half);

  v8f o[4];
  float mrow[8], lrow[8];
#pragma unroll
  for (int t = 0; t < 4; ++t) o[t] = {0, 0, 0, 0, 0, 0, 0, 0};
#pragma unroll
  for (int r = 0; r < 8; ++r) { mrow[r] = -1e30f; lrow[r] = 0.0f; }

  const float scale = 0.125f;  // 1/sqrt(dk)
  unsigned short* Pw = &Pl[wv * 16 * 64];

  for (int kb0 = 0; kb0 < S_LEN; kb0 += 64) {
    __syncthreads();
    // stage K block [key][dk] and V^T block [dv][key]: 512 b128 chunks each
#pragma unroll
    for (int i = 0; i < 4; ++i) {
      int fid = tid + i * 128;
      int row = fid >> 3, c = fid & 7;
      async_copy_b128(lds_off(&Kt[row * 64 + c * 8]), Kb,
                      (unsigned)(((kb0 + row) * DMODEL + c * 8) * 2));
      async_copy_b128(lds_off(&Vt[row * 64 + c * 8]), VTb,
                      (unsigned)((row * S_LEN + kb0 + c * 8) * 2));
    }
    wait_async0();
    __syncthreads();

    // scores: four 16x16 key tiles, contraction over dk=64
    v8f s[4];
#pragma unroll
    for (int nt = 0; nt < 4; ++nt) {
      s[nt] = {0, 0, 0, 0, 0, 0, 0, 0};
      s[nt] = wmma_bf16(aq0, load_frag(Kt, nt * 16 + ln, 64, 0,  half), s[nt]);
      s[nt] = wmma_bf16(aq1, load_frag(Kt, nt * 16 + ln, 64, 32, half), s[nt]);
    }

    // online softmax per matrix row (row = r + 8*half, cols on 16 lanes)
#pragma unroll
    for (int r = 0; r < 8; ++r) {
      float x0 = s[0][r] * scale, x1 = s[1][r] * scale;
      float x2 = s[2][r] * scale, x3 = s[3][r] * scale;
      float mx = fmaxf(fmaxf(x0, x1), fmaxf(x2, x3));
      mx = fmaxf(mx, __shfl_xor(mx, 8, 32));
      mx = fmaxf(mx, __shfl_xor(mx, 4, 32));
      mx = fmaxf(mx, __shfl_xor(mx, 2, 32));
      mx = fmaxf(mx, __shfl_xor(mx, 1, 32));
      float mnew = fmaxf(mrow[r], mx);
      float fsc  = __expf(mrow[r] - mnew);
      float p0 = __expf(x0 - mnew), p1 = __expf(x1 - mnew);
      float p2 = __expf(x2 - mnew), p3 = __expf(x3 - mnew);
      float sum = (p0 + p1) + (p2 + p3);
      sum += __shfl_xor(sum, 8, 32);
      sum += __shfl_xor(sum, 4, 32);
      sum += __shfl_xor(sum, 2, 32);
      sum += __shfl_xor(sum, 1, 32);
      lrow[r] = lrow[r] * fsc + sum;
      mrow[r] = mnew;
#pragma unroll
      for (int t = 0; t < 4; ++t) o[t][r] *= fsc;
      int prow = r + 8 * half;
      Pw[prow * 64 + ln]      = f2bf(p0);
      Pw[prow * 64 + 16 + ln] = f2bf(p1);
      Pw[prow * 64 + 32 + ln] = f2bf(p2);
      Pw[prow * 64 + 48 + ln] = f2bf(p3);
    }
    asm volatile("s_wait_dscnt 0x0" ::: "memory");

    v16bf ap0 = load_frag(Pw, ln, 64, 0,  half);
    v16bf ap1 = load_frag(Pw, ln, 64, 32, half);
#pragma unroll
    for (int t = 0; t < 4; ++t) {
      o[t] = wmma_bf16(ap0, load_frag(Vt, t * 16 + ln, 64, 0,  half), o[t]);
      o[t] = wmma_bf16(ap1, load_frag(Vt, t * 16 + ln, 64, 32, half), o[t]);
    }
  }

  // normalize and store f32 output
  float* Ob = out + head_off;
#pragma unroll
  for (int r = 0; r < 8; ++r) {
    float inv = 1.0f / lrow[r];
    int m = r + 8 * half;
#pragma unroll
    for (int t = 0; t < 4; ++t)
      Ob[(size_t)(qbase + wv * 16 + m) * DMODEL + t * 16 + ln] = o[t][r] * inv;
  }
}

// ---------------------------------------------------------------------------
extern "C" void kernel_launch(void* const* d_in, const int* in_sizes, int n_in,
                              void* d_out, int out_size, void* d_ws, size_t ws_size,
                              hipStream_t stream) {
  (void)in_sizes; (void)n_in; (void)out_size; (void)ws_size;
  const float* q   = (const float*)d_in[0];
  const float* k   = (const float*)d_in[1];
  const float* v   = (const float*)d_in[2];
  const float* wq  = (const float*)d_in[3];
  const float* bq  = (const float*)d_in[4];
  const float* wk  = (const float*)d_in[5];
  const float* bk  = (const float*)d_in[6];
  const float* wv_ = (const float*)d_in[7];
  const float* bv  = (const float*)d_in[8];

  const size_t tok = (size_t)8 * S_LEN;            // 8192 token rows
  const size_t wsz = (size_t)DMODEL * DMODEL;      // 1M weight elements
  unsigned short* Qw  = (unsigned short*)d_ws;
  unsigned short* Kw  = Qw + tok * DMODEL;
  unsigned short* VTw = Kw + tok * DMODEL;
  unsigned short* WqT = VTw + tok * DMODEL;
  unsigned short* WkT = WqT + wsz;
  unsigned short* WvT = WkT + wsz;

  dim3 tgrid(DMODEL / 32, DMODEL / 32);
  wprep_kernel<<<tgrid, 256, 0, stream>>>(wq, WqT);
  wprep_kernel<<<tgrid, 256, 0, stream>>>(wk, WkT);
  wprep_kernel<<<tgrid, 256, 0, stream>>>(wv_, WvT);

  dim3 pgrid(DMODEL / 256, (unsigned)(tok / 64));
  proj_bf16_kernel<<<pgrid, 256, 0, stream>>>(q, WqT, bq, Qw, 0);
  proj_bf16_kernel<<<pgrid, 256, 0, stream>>>(k, WkT, bk, Kw, 0);
  proj_bf16_kernel<<<pgrid, 256, 0, stream>>>(v, WvT, bv, VTw, 1);

  dim3 agrid(S_LEN / 64, NHEAD, 8);
  flash_attn_kernel<<<agrid, 128, 0, stream>>>(Qw, Kw, VTw, (float*)d_out);
}
